// DiffHead_38757784879910
// MI455X (gfx1250) — matrix operand order repaired and checked by reading, not compiled, not run on hardware
//
#include <hip/hip_runtime.h>
#include <stdint.h>
#include <stddef.h>

// ---------------- model constants ----------------
#define N_TOK   9216          // (BNW*NL)*T tokens for the 1x1 GEMMs
#define EPS_BN  1e-5f

// setup_inputs() flattening order: x0,x1,x2,nw then params dict in insertion order
enum {
  IN_X0 = 0, IN_X1, IN_X2, IN_NW,
  P_T_W, P_T_B, P_MX_PRE_G, P_MX_PRE_B,
  P_MX_W0, P_MX_B0, P_MX_G0, P_MX_BB0,
  P_MX_W1, P_MX_B1, P_MX_G1, P_MX_BB1,
  P_ENC_DW_W, P_ENC_DW_B, P_ENC_PW_W, P_ENC_PW_B, P_ENC_G, P_ENC_B,
  P_STEM_W, P_STEM_B, P_STEM_G, P_STEM_BB,
  P_BLK_W1, P_BLK_B1, P_BLK_G1, P_BLK_BB1,
  P_BLK_W2, P_BLK_B2, P_BLK_G2, P_BLK_BB2,
  P_SE1_W1, P_SE1_W2, P_SE2_W1, P_SE2_W2,
  P_FUSE_W, P_FUSE_B
};

typedef __attribute__((ext_vector_type(16))) __bf16           v16bf;
typedef __attribute__((ext_vector_type(8)))  float            v8f;
typedef __attribute__((ext_vector_type(16))) unsigned short   v16u;
typedef __attribute__((ext_vector_type(8)))  unsigned short   v8u;

__device__ __forceinline__ unsigned short f2bf(float f) {
  unsigned int u = __float_as_uint(f);
  u += 0x7fffu + ((u >> 16) & 1u);          // round-to-nearest-even
  return (unsigned short)(u >> 16);
}
__device__ __forceinline__ float bf2f(unsigned short h) {
  return __uint_as_float(((unsigned int)h) << 16);
}
__device__ __forceinline__ float bnsc(float g) { return g * rsqrtf(1.0f + EPS_BN); }
__device__ __forceinline__ float gelu_f(float x) {
  return 0.5f * x * (1.0f + erff(x * 0.70710678118654752f));
}

// ---- CDNA5 async copy: global -> LDS, per-lane 8 bytes, tracked by ASYNCcnt ----
// LDS aperture generic addresses carry the wave-relative LDS byte offset in
// addr[31:0] (ISA 00_overview §10.2 aperture table), so the VDST operand is
// just the low 32 bits of the generic pointer.
__device__ __forceinline__ unsigned lds_off(const void* p) {
  return (unsigned)(size_t)p;
}
__device__ __forceinline__ void async_b64(unsigned dst_lds, const void* gsrc) {
  asm volatile("global_load_async_to_lds_b64 %0, %1, off"
               :: "v"(dst_lds), "v"(gsrc) : "memory");
}
__device__ __forceinline__ void wait_async0() {
#if __has_builtin(__builtin_amdgcn_s_wait_asynccnt)
  __builtin_amdgcn_s_wait_asynccnt(0);
#else
  asm volatile("s_wait_asynccnt 0" ::: "memory");
#endif
}

// ---- WMMA fragment loads from LDS (contiguous k order k = 16*(lane/16) + s, same for A and B) ----
// A (16x32 bf16), LDS row-major [m][32]:
//   lanes 0-15 hold row m=lane, K 0..15; lanes 16-31 K 16..31
__device__ __forceinline__ v16bf load_afrag(const unsigned short* As, int mrow, int lane) {
  const v8u* p = (const v8u*)(As + mrow * 32);
  int kh = lane >> 4;            // 0 or 1
  v8u lo = p[2 * kh];            // K = 16*kh .. 16*kh+7
  v8u hi = p[2 * kh + 1];        // K = 16*kh+8 .. +15
  v16u t;
#pragma unroll
  for (int j = 0; j < 8; ++j) { t[j] = lo[j]; t[j + 8] = hi[j]; }
  return __builtin_bit_cast(v16bf, t);
}
// B (32x16 bf16), LDS stored column-major [col][32]:
//   lanes 0-15 hold col n=lane, K 0..15; lanes 16-31 hold K 16..31
__device__ __forceinline__ v16bf load_bfrag(const unsigned short* Bs, int col, int lane) {
  const v16u* p = (const v16u*)(Bs + col * 32 + ((lane >> 4) << 4));
  return __builtin_bit_cast(v16bf, *p);
}

#define WMMA_BF16(A, B, C) \
  __builtin_amdgcn_wmma_f32_16x16x32_bf16(false, (A), false, (B), (short)0, (C), false, false)

// =====================================================================
// weight pre-conversion: f32 -> bf16 (so GEMM tiles can be async-copied)
// =====================================================================
__global__ void k_cvt_bf16(const float* __restrict__ src, unsigned short* __restrict__ dst,
                           int n) {
  int i = blockIdx.x * 256 + threadIdx.x;
  if (i < n) dst[i] = f2bf(src[i]);
}
// conv weights (co,ci,tap) -> bf16 [co][tap][ci] so each (co,tap) row of 256 ci
// is contiguous
__global__ void k_cvt_wconv(const float* __restrict__ src, unsigned short* __restrict__ dst) {
  int i = blockIdx.x * 256 + threadIdx.x;   // 256*256*9 exact
  int tap = i % 9;
  int ci = (i / 9) % 256;
  int co = i / (9 * 256);
  dst[(co * 9 + tap) * 256 + ci] = f2bf(src[i]);
}

// =====================================================================
// K1: temporal depthwise conv1d(k=3,pad=1) + ReLU, feat = 0.5*(t + x)
// =====================================================================
__global__ __launch_bounds__(256) void k_temporal(
    const float* __restrict__ x0, const float* __restrict__ x1, const float* __restrict__ x2,
    const float* __restrict__ tw, const float* __restrict__ tb,
    float* __restrict__ outf) {
  int idx = blockIdx.x * 256 + threadIdx.x;        // 32*3*256*96 exact
  int t = idx % 96;
  int c = (idx / 96) % 256;
  int l = (idx / (96 * 256)) % 3;
  int b = idx / (96 * 256 * 3);
  const float* x = (l == 0) ? x0 : ((l == 1) ? x1 : x2);
  int base = (b * 256 + c) * 96;
  float s = tb[l * 256 + c];
#pragma unroll
  for (int j = 0; j < 3; ++j) {
    int tt = t + j - 1;
    if ((unsigned)tt < 96u) s += tw[(l * 256 + c) * 3 + j] * x[base + tt];
  }
  s = fmaxf(s, 0.0f);
  outf[idx] = 0.5f * (s + x[base + t]);
}

// =====================================================================
// K2: mixer staging: xn = bn(x); dp = diff_pad(xn); stage bf16 token-major
//     [token][c] (contiguous 32-channel runs for async tile copies); dm = x
// =====================================================================
__global__ __launch_bounds__(256) void k_mixer_stage(
    const float* __restrict__ outf /* viewed (96,256,96) */,
    const float* __restrict__ g, const float* __restrict__ bt,
    unsigned short* __restrict__ xn_st, unsigned short* __restrict__ dp_st,
    float* __restrict__ dm) {
  int idx = blockIdx.x * 256 + threadIdx.x;        // 96*256*96 exact
  int t = idx % 96;
  int c = (idx / 96) % 256;
  int nb = idx / (96 * 256);
  float x = outf[idx];
  float sc = bnsc(g[c]);
  float xn = x * sc + bt[c];
  int t1 = (t == 0) ? 1 : t;                       // replicate-pad diff
  float xa = outf[idx - t + t1];
  float xb = outf[idx - t + t1 - 1];
  float dp = (xa - xb) * sc;                       // bn bias cancels in the diff
  int n = nb * 96 + t;
  xn_st[n * 256 + c] = f2bf(xn);
  dp_st[n * 256 + c] = f2bf(dp);
  dm[idx] = x;
}

// =====================================================================
// K3: 1x1-conv GEMM via WMMA + async LDS staging.
// A: bf16 weights [co][256]; B: bf16 activations [token][256].
// grid (4, 96): co-tile 64, token-tile 96.  mode 0: dm += gelu(bn(acc+bias))
// mode 1: out = relu(bn(acc+bias))
// =====================================================================
__global__ __launch_bounds__(256) void k_gemm256_wmma(
    const unsigned short* __restrict__ B /* [9216][256] bf16 */,
    const unsigned short* __restrict__ Wbf /* [256][256] bf16 */,
    const float* __restrict__ bias, const float* __restrict__ g, const float* __restrict__ bt,
    float* __restrict__ outbuf, int mode) {
  __shared__ __align__(128) unsigned short As[64 * 32];
  __shared__ __align__(128) unsigned short Bs[96 * 32];
  const int co0 = blockIdx.x * 64;
  const int nb = blockIdx.y;
  const int tid = threadIdx.x;
  const int lane = tid & 31, wave = tid >> 5;
  const int wm = wave & 3, wn = wave >> 2;
  const unsigned asb = lds_off(&As[0]);
  const unsigned bsb = lds_off(&Bs[0]);
  v8f acc0 = {}, acc1 = {}, acc2 = {};
  for (int k0 = 0; k0 < 256; k0 += 32) {
#pragma unroll
    for (int c = tid; c < 64 * 8; c += 256) {      // A tile: 64 rows x 64B
      int m = c >> 3, j = c & 7;
      async_b64(asb + m * 64 + j * 8, Wbf + (co0 + m) * 256 + k0 + j * 4);
    }
#pragma unroll
    for (int c = tid; c < 96 * 8; c += 256) {      // B tile: 96 rows x 64B
      int x = c >> 3, j = c & 7;
      async_b64(bsb + x * 64 + j * 8, B + (size_t)(nb * 96 + x) * 256 + k0 + j * 4);
    }
    __builtin_prefetch(Wbf + (co0 + (tid >> 2)) * 256 + k0 + 32, 0, 1);
    wait_async0();
    __syncthreads();
    v16bf a = load_afrag(As, wm * 16 + (lane & 15), lane);
    v16bf b0 = load_bfrag(Bs, wn * 48 + 0 + (lane & 15), lane);
    v16bf b1 = load_bfrag(Bs, wn * 48 + 16 + (lane & 15), lane);
    v16bf b2 = load_bfrag(Bs, wn * 48 + 32 + (lane & 15), lane);
    acc0 = WMMA_BF16(a, b0, acc0);
    acc1 = WMMA_BF16(a, b1, acc1);
    acc2 = WMMA_BF16(a, b2, acc2);
    __syncthreads();
  }
#pragma unroll
  for (int nt = 0; nt < 3; ++nt) {
    v8f acc = (nt == 0) ? acc0 : ((nt == 1) ? acc1 : acc2);
    int t = wn * 48 + nt * 16 + (lane & 15);
#pragma unroll
    for (int r = 0; r < 8; ++r) {
      int co = co0 + wm * 16 + r + ((lane >> 4) << 3);  // C layout: M = r + 8*(lane/16)
      float v = acc[r] + bias[co];
      v = v * bnsc(g[co]) + bt[co];
      int idx = (nb * 256 + co) * 96 + t;
      if (mode == 0) outbuf[idx] += gelu_f(v);
      else           outbuf[idx] = fmaxf(v, 0.0f);
    }
  }
}

// =====================================================================
// K4: encoder depthwise conv1d(k=3,pad=1)+bias, stage bf16 [token][c]
// =====================================================================
__global__ __launch_bounds__(256) void k_enc_dw(
    const float* __restrict__ in /* (96,256,96) */,
    const float* __restrict__ w /* [256][3] */, const float* __restrict__ bias,
    unsigned short* __restrict__ st) {
  int idx = blockIdx.x * 256 + threadIdx.x;        // 96*256*96 exact
  int t = idx % 96;
  int c = (idx / 96) % 256;
  int nb = idx / (96 * 256);
  float s = bias[c];
#pragma unroll
  for (int j = 0; j < 3; ++j) {
    int tt = t + j - 1;
    if ((unsigned)tt < 96u) s += w[c * 3 + j] * in[idx - t + tt];
  }
  st[(nb * 96 + t) * 256 + c] = f2bf(s);
}

// =====================================================================
// K5: cosine-similarity Gram per (b,l): dmi[b][l][t][s]
// =====================================================================
__global__ __launch_bounds__(256) void k_cosine(
    const float* __restrict__ outf, const float* __restrict__ dm,
    float* __restrict__ dmi) {
  __shared__ __align__(16) unsigned short X[256 * 96];  // [c][t] bf16, 48 KB
  __shared__ float nrm[96];
  int l = blockIdx.x, b = blockIdx.y, tid = threadIdx.x;
  const float* src = (l < 3) ? (outf + (size_t)((b * 3 + l) * 256) * 96)
                             : (dm + (size_t)((b * 3 + (l - 3)) * 256) * 96);
  for (int i = tid; i < 256 * 96; i += 256) X[i] = f2bf(src[i]);
  __syncthreads();
  if (tid < 96) {
    float s = 0.0f;
    for (int c = 0; c < 256; ++c) { float v = bf2f(X[c * 96 + tid]); s += v * v; }
    nrm[tid] = sqrtf(s);
  }
  __syncthreads();
  for (int p = tid; p < 96 * 96; p += 256) {
    int t = p / 96, s = p % 96;
    float d = 0.0f;
    for (int c = 0; c < 256; ++c) d += bf2f(X[c * 96 + t]) * bf2f(X[c * 96 + s]);
    float den = fmaxf(nrm[t] * nrm[s], 1e-8f);
    dmi[(size_t)((b * 6 + l) * 96 + t) * 96 + s] = d / den;
  }
}

// =====================================================================
// K6: stem conv2d 6->256 3x3 pad1 + BN + ReLU -> bf16 NHWC [b][y][x][co]
// =====================================================================
__global__ __launch_bounds__(256) void k_stem(
    const float* __restrict__ dmi /* (32,6,96,96) */,
    const float* __restrict__ w /* (256,6,3,3) */, const float* __restrict__ bias,
    const float* __restrict__ g, const float* __restrict__ bt,
    unsigned short* __restrict__ h) {
  size_t idx = (size_t)blockIdx.x * 256 + threadIdx.x;   // 32*96*96*256 exact
  int co = (int)(idx & 255);
  int sp = (int)(idx >> 8);
  int x = sp % 96;
  int y = (sp / 96) % 96;
  int b = sp / (96 * 96);
  float s = bias[co];
#pragma unroll
  for (int ci = 0; ci < 6; ++ci)
#pragma unroll
    for (int ky = 0; ky < 3; ++ky) {
      int yy = y + ky - 1;
      if ((unsigned)yy >= 96u) continue;
#pragma unroll
      for (int kx = 0; kx < 3; ++kx) {
        int xx = x + kx - 1;
        if ((unsigned)xx >= 96u) continue;
        s += w[((co * 6 + ci) * 3 + ky) * 3 + kx] *
             dmi[(size_t)((b * 6 + ci) * 96 + yy) * 96 + xx];
      }
    }
  s = fmaxf(s * bnsc(g[co]) + bt[co], 0.0f);
  h[idx] = f2bf(s);   // idx == ((b*96+y)*96+x)*256+co
}

// =====================================================================
// K7: heavy 3x3 conv 256->256, implicit GEMM via WMMA + async LDS staging.
// Input NHWC bf16 [b][y][x][ci]; weights bf16 [co][tap][ci].
// Block: 256 thr (8 waves), tile M=128 (co) x N=96 (one output row).
// grid (2, 96, 32).  K loop: 9 taps x 8 ci-chunks of 32 (OOB rows skipped).
// mode 0: out NHWC bf16 = relu(bn1(acc+b1))
// mode 1: v = relu(h + bn2(acc+b2)); atomicAdd(dmap[b][co][y], v/96)
// =====================================================================
__global__ __launch_bounds__(256) void k_conv3x3_wmma(
    const unsigned short* __restrict__ in /* [32][96][96][256] bf16 */,
    const unsigned short* __restrict__ Wbf /* [256][9][256] bf16 */,
    const float* __restrict__ bias, const float* __restrict__ g, const float* __restrict__ bt,
    unsigned short* __restrict__ outbf,
    const unsigned short* __restrict__ hres,
    float* __restrict__ dmap, int mode) {
  __shared__ __align__(128) unsigned short As[128 * 32];   // 8 KB
  __shared__ __align__(128) unsigned short Bs[96 * 32];    // 6 KB
  const int co0 = blockIdx.x * 128;
  const int y = blockIdx.y;
  const int b = blockIdx.z;
  const int tid = threadIdx.x;
  const int lane = tid & 31, wave = tid >> 5;
  const int wm = wave & 3, wn = wave >> 2;   // wm: 32-row slice, wn: 48-col half
  const unsigned asb = lds_off(&As[0]);
  const unsigned bsb = lds_off(&Bs[0]);
  v8f acc[6];
#pragma unroll
  for (int i = 0; i < 6; ++i) acc[i] = (v8f){};
  for (int tap = 0; tap < 9; ++tap) {
    const int dy = tap / 3 - 1, dx = tap % 3 - 1;
    const int yy = y + dy;
    if ((unsigned)yy >= 96u) continue;       // zero contribution rows: skip
    for (int ci0 = 0; ci0 < 256; ci0 += 32) {
#pragma unroll
      for (int c = tid; c < 128 * 8; c += 256) {   // A tile: 128 rows x 64B
        int m = c >> 3, j = c & 7;
        async_b64(asb + m * 64 + j * 8,
                  Wbf + ((size_t)(co0 + m) * 9 + tap) * 256 + ci0 + j * 4);
      }
#pragma unroll
      for (int c = tid; c < 96 * 8; c += 256) {    // B tile: 96 rows x 64B
        int x = c >> 3, j = c & 7;
        int xx = x + dx;
        if ((unsigned)xx < 96u) {
          async_b64(bsb + x * 64 + j * 8,
                    in + ((size_t)((b * 96 + yy) * 96 + xx)) * 256 + ci0 + j * 4);
        } else {
          *(unsigned long long*)&Bs[x * 32 + j * 4] = 0ull;   // pad column
        }
      }
      __builtin_prefetch(Wbf + ((size_t)(co0 + tid) * 9 + tap) * 256 + ci0 + 32, 0, 1);
      wait_async0();
      __syncthreads();
      v16bf a0 = load_afrag(As, wm * 32 + (lane & 15), lane);
      v16bf a1 = load_afrag(As, wm * 32 + 16 + (lane & 15), lane);
      v16bf b0 = load_bfrag(Bs, wn * 48 + 0 + (lane & 15), lane);
      v16bf b1 = load_bfrag(Bs, wn * 48 + 16 + (lane & 15), lane);
      v16bf b2 = load_bfrag(Bs, wn * 48 + 32 + (lane & 15), lane);
      acc[0] = WMMA_BF16(a0, b0, acc[0]);
      acc[1] = WMMA_BF16(a0, b1, acc[1]);
      acc[2] = WMMA_BF16(a0, b2, acc[2]);
      acc[3] = WMMA_BF16(a1, b0, acc[3]);
      acc[4] = WMMA_BF16(a1, b1, acc[4]);
      acc[5] = WMMA_BF16(a1, b2, acc[5]);
      __syncthreads();
    }
  }
#pragma unroll
  for (int sub = 0; sub < 2; ++sub)
#pragma unroll
    for (int nt = 0; nt < 3; ++nt) {
      v8f a = acc[sub * 3 + nt];
      int x = wn * 48 + nt * 16 + (lane & 15);
#pragma unroll
      for (int r = 0; r < 8; ++r) {
        int co = co0 + wm * 32 + sub * 16 + r + ((lane >> 4) << 3);
        float v = a[r] + bias[co];
        v = v * bnsc(g[co]) + bt[co];
        size_t oidx = ((size_t)((b * 96 + y) * 96 + x)) * 256 + co;
        if (mode == 0) {
          outbf[oidx] = f2bf(fmaxf(v, 0.0f));
        } else {
          float hv = bf2f(hres[oidx]);
          v = fmaxf(hv + v, 0.0f);
          atomicAdd(&dmap[((b << 8) + co) * 96 + y], v * (1.0f / 96.0f));
        }
      }
    }
}

// =====================================================================
// K8: zero fill (for dmap accumulator)
// =====================================================================
__global__ void k_zero(float* __restrict__ p, int n) {
  int i = blockIdx.x * 256 + threadIdx.x;
  if (i < n) p[i] = 0.0f;
}

// =====================================================================
// K9: d1d = mean over NL of final diff_1d  (32,256,96)
// =====================================================================
__global__ __launch_bounds__(256) void k_d1d(
    const float* __restrict__ enc /* (96,256,96) */, float* __restrict__ d1d) {
  int idx = blockIdx.x * 256 + threadIdx.x;        // 32*256*96 exact
  int t = idx % 96;
  int c = (idx / 96) % 256;
  int b = idx / (96 * 256);
  float s = 0.0f;
#pragma unroll
  for (int l = 0; l < 3; ++l) s += enc[((b * 3 + l) * 256 + c) * 96 + t];
  d1d[idx] = s * (1.0f / 3.0f);
}

// =====================================================================
// K10: SE heads + fuse.  One block per b.  Output (32,256) == d_out flat.
// =====================================================================
__global__ __launch_bounds__(256) void k_head(
    const float* __restrict__ d1d, const float* __restrict__ dmap,
    const float* __restrict__ se1w1, const float* __restrict__ se1w2,
    const float* __restrict__ se2w1, const float* __restrict__ se2w2,
    const float* __restrict__ fw, const float* __restrict__ fb,
    float* __restrict__ out) {
  __shared__ float p1[256], p2[256], h1[64], h2[64], hcat[512];
  int b = blockIdx.x, c = threadIdx.x;
  float s = 0.0f;
  for (int t = 0; t < 96; ++t) s += d1d[(b * 256 + c) * 96 + t];
  p1[c] = s * (1.0f / 96.0f);
  s = 0.0f;
  for (int y = 0; y < 96; ++y) s += dmap[(b * 256 + c) * 96 + y];
  p2[c] = s * (1.0f / 96.0f);
  __syncthreads();
  if (c < 64) {
    float a = 0.0f, bb = 0.0f;
    for (int k = 0; k < 256; ++k) {
      a += se1w1[c * 256 + k] * p1[k];
      bb += se2w1[c * 256 + k] * p2[k];
    }
    h1[c] = fmaxf(a, 0.0f);
    h2[c] = fmaxf(bb, 0.0f);
  }
  __syncthreads();
  float a1 = 0.0f, a2 = 0.0f;
  for (int k = 0; k < 64; ++k) {
    a1 += se1w2[c * 64 + k] * h1[k];
    a2 += se2w2[c * 64 + k] * h2[k];
  }
  float s1 = 1.0f / (1.0f + expf(-a1));
  float s2 = 1.0f / (1.0f + expf(-a2));
  hcat[c] = p2[c] * s1;          // img2ddm = s1 * mean_y(diff_map)
  hcat[256 + c] = p1[c] * s2;    // ddm2img = s2 * mean_t(d1d)
  __syncthreads();
  float o = fb[c];
  for (int k = 0; k < 512; ++k) o += fw[c * 512 + k] * hcat[k];
  out[b * 256 + c] = fmaxf(o, 0.0f);
}

// =====================================================================
// host
// =====================================================================
static inline char* carve(char*& p, size_t bytes) {
  char* r = p;
  p += (bytes + 255) & ~(size_t)255;
  return r;
}

extern "C" void kernel_launch(void* const* d_in, const int* in_sizes, int n_in,
                              void* d_out, int out_size, void* d_ws, size_t ws_size,
                              hipStream_t stream) {
  (void)in_sizes; (void)n_in; (void)out_size; (void)ws_size;
  const float* x0 = (const float*)d_in[IN_X0];
  const float* x1 = (const float*)d_in[IN_X1];
  const float* x2 = (const float*)d_in[IN_X2];
  const float* t_w = (const float*)d_in[P_T_W];
  const float* t_b = (const float*)d_in[P_T_B];
  const float* mx_pre_g = (const float*)d_in[P_MX_PRE_G];
  const float* mx_pre_b = (const float*)d_in[P_MX_PRE_B];
  const float* mx_w0 = (const float*)d_in[P_MX_W0];
  const float* mx_b0 = (const float*)d_in[P_MX_B0];
  const float* mx_g0 = (const float*)d_in[P_MX_G0];
  const float* mx_bb0 = (const float*)d_in[P_MX_BB0];
  const float* mx_w1 = (const float*)d_in[P_MX_W1];
  const float* mx_b1 = (const float*)d_in[P_MX_B1];
  const float* mx_g1 = (const float*)d_in[P_MX_G1];
  const float* mx_bb1 = (const float*)d_in[P_MX_BB1];
  const float* enc_dw_w = (const float*)d_in[P_ENC_DW_W];
  const float* enc_dw_b = (const float*)d_in[P_ENC_DW_B];
  const float* enc_pw_w = (const float*)d_in[P_ENC_PW_W];
  const float* enc_pw_b = (const float*)d_in[P_ENC_PW_B];
  const float* enc_g = (const float*)d_in[P_ENC_G];
  const float* enc_b = (const float*)d_in[P_ENC_B];
  const float* stem_w = (const float*)d_in[P_STEM_W];
  const float* stem_b = (const float*)d_in[P_STEM_B];
  const float* stem_g = (const float*)d_in[P_STEM_G];
  const float* stem_bb = (const float*)d_in[P_STEM_BB];
  const float* blk_w1 = (const float*)d_in[P_BLK_W1];
  const float* blk_b1 = (const float*)d_in[P_BLK_B1];
  const float* blk_g1 = (const float*)d_in[P_BLK_G1];
  const float* blk_bb1 = (const float*)d_in[P_BLK_BB1];
  const float* blk_w2 = (const float*)d_in[P_BLK_W2];
  const float* blk_b2 = (const float*)d_in[P_BLK_B2];
  const float* blk_g2 = (const float*)d_in[P_BLK_G2];
  const float* blk_bb2 = (const float*)d_in[P_BLK_BB2];
  const float* se1_w1 = (const float*)d_in[P_SE1_W1];
  const float* se1_w2 = (const float*)d_in[P_SE1_W2];
  const float* se2_w1 = (const float*)d_in[P_SE2_W1];
  const float* se2_w2 = (const float*)d_in[P_SE2_W2];
  const float* fuse_w = (const float*)d_in[P_FUSE_W];
  const float* fuse_b = (const float*)d_in[P_FUSE_B];
  float* out = (float*)d_out;

  // workspace carving (~355 MB total)
  char* ws = (char*)d_ws;
  float* outf = (float*)carve(ws, (size_t)32 * 3 * 256 * 96 * 4);          // temporal feats
  float* dm   = (float*)carve(ws, (size_t)96 * 256 * 96 * 4);              // mixer output
  unsigned short* st1 = (unsigned short*)carve(ws, (size_t)N_TOK * 256 * 2);
  unsigned short* st2 = (unsigned short*)carve(ws, (size_t)N_TOK * 256 * 2);
  float* encA = (float*)carve(ws, (size_t)96 * 256 * 96 * 4);
  float* encB = (float*)carve(ws, (size_t)96 * 256 * 96 * 4);
  float* dmi  = (float*)carve(ws, (size_t)32 * 6 * 96 * 96 * 4);
  unsigned short* hbf = (unsigned short*)carve(ws, (size_t)32 * 96 * 96 * 256 * 2); // stem (NHWC)
  unsigned short* rbf = (unsigned short*)carve(ws, (size_t)32 * 96 * 96 * 256 * 2); // conv1 (NHWC)
  float* dmap = (float*)carve(ws, (size_t)32 * 256 * 96 * 4);
  float* d1d  = (float*)carve(ws, (size_t)32 * 256 * 96 * 4);
  unsigned short* w1x0 = (unsigned short*)carve(ws, (size_t)256 * 256 * 2);  // bf16 1x1 weights
  unsigned short* w1x1 = (unsigned short*)carve(ws, (size_t)256 * 256 * 2);
  unsigned short* wpw0 = (unsigned short*)carve(ws, (size_t)256 * 256 * 2);
  unsigned short* wpw1 = (unsigned short*)carve(ws, (size_t)256 * 256 * 2);
  unsigned short* wc1 = (unsigned short*)carve(ws, (size_t)256 * 9 * 256 * 2); // bf16 conv wts
  unsigned short* wc2 = (unsigned short*)carve(ws, (size_t)256 * 9 * 256 * 2);

  // 0. weight pre-conversion to bf16 (async-copyable layouts)
  k_cvt_bf16<<<256, 256, 0, stream>>>(mx_w0, w1x0, 256 * 256);
  k_cvt_bf16<<<256, 256, 0, stream>>>(mx_w1, w1x1, 256 * 256);
  k_cvt_bf16<<<256, 256, 0, stream>>>(enc_pw_w, wpw0, 256 * 256);
  k_cvt_bf16<<<256, 256, 0, stream>>>(enc_pw_w + 256 * 256, wpw1, 256 * 256);
  k_cvt_wconv<<<2304, 256, 0, stream>>>(blk_w1, wc1);
  k_cvt_wconv<<<2304, 256, 0, stream>>>(blk_w2, wc2);

  // 1. temporal feats
  k_temporal<<<9216, 256, 0, stream>>>(x0, x1, x2, t_w, t_b, outf);
  // 2. mixer staging (xn, diff-pad, dm = x)
  k_mixer_stage<<<9216, 256, 0, stream>>>(outf, mx_pre_g, mx_pre_b, st1, st2, dm);
  // 3-4. mixer 1x1 GEMMs, accumulate gelu(bn(.)) into dm
  k_gemm256_wmma<<<dim3(4, 96), 256, 0, stream>>>(st1, w1x0, mx_b0, mx_g0, mx_bb0, dm, 0);
  k_gemm256_wmma<<<dim3(4, 96), 256, 0, stream>>>(st2, w1x1, mx_b1, mx_g1, mx_bb1, dm, 0);
  // 5-8. encoder: dw -> pw(WMMA)+bn+relu, twice
  k_enc_dw<<<9216, 256, 0, stream>>>(dm, enc_dw_w, enc_dw_b, st1);
  k_gemm256_wmma<<<dim3(4, 96), 256, 0, stream>>>(st1, wpw0, enc_pw_b, enc_g, enc_b, encA, 1);
  k_enc_dw<<<9216, 256, 0, stream>>>(encA, enc_dw_w + 256 * 3, enc_dw_b + 256, st1);
  k_gemm256_wmma<<<dim3(4, 96), 256, 0, stream>>>(st1, wpw1, enc_pw_b + 256,
                                                  enc_g + 256, enc_b + 256, encB, 1);
  // 9. cosine similarity maps
  k_cosine<<<dim3(6, 32), 256, 0, stream>>>(outf, dm, dmi);
  // 10. stem conv (Cin=6) + bn + relu -> bf16 NHWC
  k_stem<<<294912, 256, 0, stream>>>(dmi, stem_w, stem_b, stem_g, stem_bb, hbf);
  // 11. residual block conv1 (WMMA, async staging) -> bf16 NHWC
  k_conv3x3_wmma<<<dim3(2, 96, 32), 256, 0, stream>>>(hbf, wc1, blk_b1, blk_g1, blk_bb1,
                                                      rbf, nullptr, nullptr, 0);
  // 12. zero diff_map accumulator, then conv2 fused residual+relu+mean
  k_zero<<<3072, 256, 0, stream>>>(dmap, 32 * 256 * 96);
  k_conv3x3_wmma<<<dim3(2, 96, 32), 256, 0, stream>>>(rbf, wc2, blk_b2, blk_g2, blk_bb2,
                                                      nullptr, hbf, dmap, 1);
  // 13. d1d = mean over NL
  k_d1d<<<3072, 256, 0, stream>>>(encB, d1d);
  // 14. SE + fuse head -> d_out
  k_head<<<32, 256, 0, stream>>>(d1d, dmap, se1_w1, se1_w2, se2_w1, se2_w2, fuse_w, fuse_b, out);
}
